// MoE_40999757807741
// MI455X (gfx1250) — matrix operand order repaired and checked by reading, not compiled, not run on hardware
//
#include <hip/hip_runtime.h>

#define T_TOK 16384
#define D_DIM 1024
#define E_EXP 8
#define DFF_DIM 4096
#define C_CAP 2048
#define NBLK 64   // T_TOK / 256

typedef __attribute__((ext_vector_type(16))) __bf16 v16bf;
typedef __attribute__((ext_vector_type(8)))  __bf16 v8bf;
typedef __attribute__((ext_vector_type(8)))  float  v8f;

__device__ __forceinline__ unsigned short f2bfu(float f) {
  unsigned u = __builtin_bit_cast(unsigned, f);
  return (unsigned short)((u + 0x7FFFu + ((u >> 16) & 1u)) >> 16);  // RNE
}
__device__ __forceinline__ __bf16 f2bf(float f) {
  unsigned short s = f2bfu(f);
  return __builtin_bit_cast(__bf16, s);
}

// tanh-approx GELU with fast exp (single v_exp_f32 instead of branchy tanhf)
__device__ __forceinline__ float gelu_tanh(float x) {
  float u = 0.7978845608028654f * (x + 0.044715f * x * x * x);
  float t = __expf(-2.0f * fabsf(u));
  float th = (1.0f - t) / (1.0f + t);
  th = copysignf(th, u);
  return 0.5f * x * (1.0f + th);
}

// ---------------------------------------------------------------- fills
__global__ void moe455_fill_u32(unsigned* __restrict__ p, unsigned v, long n) {
  long i = (long)blockIdx.x * blockDim.x + threadIdx.x;
  long st = (long)gridDim.x * blockDim.x;
  for (; i < n; i += st) p[i] = v;
}

// ---------------------------------------------------------------- weight convert + transpose
// in: W [E][KD][ND] fp32 ; out: WT [E][ND][KD] bf16  (K-major rows for GEMM B tiles)
template <int KD, int ND>
__global__ __launch_bounds__(256)
void moe455_wcvt(const float* __restrict__ W, __bf16* __restrict__ WT) {
  __shared__ float tile[64][65];
  int e = blockIdx.z;
  int kb = blockIdx.y * 64, nb = blockIdx.x * 64;
  const float* src = W + (size_t)e * KD * ND;
  __bf16* dst = WT + (size_t)e * ND * KD;
  // read 64(k) x 64(n) tile, coalesced rows
  int r = threadIdx.x >> 2, cb = (threadIdx.x & 3) * 16;
  const float4* s4 = reinterpret_cast<const float4*>(src + (size_t)(kb + r) * ND + nb + cb);
#pragma unroll
  for (int q = 0; q < 4; ++q) {
    float4 f = s4[q];
    tile[r][cb + q * 4 + 0] = f.x;
    tile[r][cb + q * 4 + 1] = f.y;
    tile[r][cb + q * 4 + 2] = f.z;
    tile[r][cb + q * 4 + 3] = f.w;
  }
  __syncthreads();
  // write transposed: row n, 16 consecutive k, packed 2x bf16 per dword
  int n = threadIdx.x >> 2, kc = (threadIdx.x & 3) * 16;
  unsigned w[8];
#pragma unroll
  for (int q = 0; q < 8; ++q)
    w[q] = (unsigned)f2bfu(tile[kc + 2 * q][n]) |
           ((unsigned)f2bfu(tile[kc + 2 * q + 1][n]) << 16);
  uint4* d4 = reinterpret_cast<uint4*>(dst + (size_t)(nb + n) * KD + kb + kc);
  d4[0] = make_uint4(w[0], w[1], w[2], w[3]);
  d4[1] = make_uint4(w[4], w[5], w[6], w[7]);
}

// ---------------------------------------------------------------- gating
__global__ __launch_bounds__(256)
void moe455_gate(const float* __restrict__ x, const float* __restrict__ wg,
                 int* __restrict__ idxArr, float* __restrict__ gvArr,
                 float* __restrict__ gsum) {
  __shared__ float ls[E_EXP];
  if (threadIdx.x < E_EXP) ls[threadIdx.x] = 0.f;
  __syncthreads();
  int wid = threadIdx.x >> 5, lane = threadIdx.x & 31;
  int t = blockIdx.x * 8 + wid;
  const float* xr = x + (size_t)t * D_DIM;
  float acc[E_EXP];
#pragma unroll
  for (int e = 0; e < E_EXP; ++e) acc[e] = 0.f;
  for (int i = 0; i < D_DIM / 32; ++i) {
    int d = lane + i * 32;
    float xv = xr[d];
    const float4* w4 = reinterpret_cast<const float4*>(wg + (size_t)d * E_EXP);
    float4 wa = w4[0], wb = w4[1];
    acc[0] += xv * wa.x; acc[1] += xv * wa.y; acc[2] += xv * wa.z; acc[3] += xv * wa.w;
    acc[4] += xv * wb.x; acc[5] += xv * wb.y; acc[6] += xv * wb.z; acc[7] += xv * wb.w;
  }
#pragma unroll
  for (int e = 0; e < E_EXP; ++e)
#pragma unroll
    for (int m = 16; m >= 1; m >>= 1) acc[e] += __shfl_xor(acc[e], m, 32);
  if (lane == 0) {
    float mx = acc[0]; int am = 0;
#pragma unroll
    for (int e = 1; e < E_EXP; ++e) if (acc[e] > mx) { mx = acc[e]; am = e; }
    float p[E_EXP]; float s = 0.f;
#pragma unroll
    for (int e = 0; e < E_EXP; ++e) { p[e] = __expf(acc[e] - mx); s += p[e]; }
    float invs = 1.f / s;
#pragma unroll
    for (int e = 0; e < E_EXP; ++e) atomicAdd(&ls[e], p[e] * invs);
    idxArr[t] = am;
    gvArr[t] = p[am] * invs;
  }
  __syncthreads();
  if (threadIdx.x < E_EXP) atomicAdd(&gsum[threadIdx.x], ls[threadIdx.x]);
}

// ---------------------------------------------------------------- block histograms
__global__ __launch_bounds__(256)
void moe455_hist(const int* __restrict__ idxArr, int* __restrict__ hist) {
  __shared__ int h[E_EXP];
  if (threadIdx.x < E_EXP) h[threadIdx.x] = 0;
  __syncthreads();
  int t = blockIdx.x * 256 + threadIdx.x;
  atomicAdd(&h[idxArr[t]], 1);
  __syncthreads();
  if (threadIdx.x < E_EXP) hist[blockIdx.x * E_EXP + threadIdx.x] = h[threadIdx.x];
}

// ---------------------------------------------------------------- scan + aux loss
__global__ void moe455_scan(const int* __restrict__ hist, int* __restrict__ boff,
                            const float* __restrict__ gsum, float* __restrict__ laux) {
  __shared__ int cnt[E_EXP];
  int e = threadIdx.x;
  if (e < E_EXP) {
    int run = 0;
    for (int b = 0; b < NBLK; ++b) { boff[b * E_EXP + e] = run; run += hist[b * E_EXP + e]; }
    cnt[e] = run;
  }
  __syncthreads();
  if (threadIdx.x == 0) {
    float l = 0.f;
    for (int q = 0; q < E_EXP; ++q)
      l += (gsum[q] / (float)T_TOK) * ((float)cnt[q] / (float)T_TOK);
    laux[0] = l * (float)E_EXP;
  }
}

// ---------------------------------------------------------------- ordered ranks / slots
__global__ __launch_bounds__(256)
void moe455_rank(const int* __restrict__ idxArr, const int* __restrict__ boff,
                 int* __restrict__ slotA, float* __restrict__ gvArr,
                 int* __restrict__ inv) {
  __shared__ int wcnt[8][E_EXP];
  __shared__ int woff[8][E_EXP];
  int wid = threadIdx.x >> 5, lane = threadIdx.x & 31;
  int t = blockIdx.x * 256 + threadIdx.x;
  int e = idxArr[t];
  unsigned myBal = 0u;
#pragma unroll
  for (int q = 0; q < E_EXP; ++q) {
    unsigned bq = (unsigned)__ballot(e == q);
    if (e == q) myBal = bq;
    if (lane == q) wcnt[wid][q] = __popc(bq);
  }
  __syncthreads();
  if (threadIdx.x < 8 * E_EXP) {
    int w = threadIdx.x >> 3, q = threadIdx.x & 7;
    int r = 0;
    for (int w2 = 0; w2 < w; ++w2) r += wcnt[w2][q];
    woff[w][q] = r;
  }
  __syncthreads();
  unsigned below = (1u << lane) - 1u;
  int rank = __popc(myBal & below);
  int pos = boff[blockIdx.x * E_EXP + e] + woff[wid][e] + rank;
  bool keep = pos < C_CAP;
  int s = keep ? pos : (C_CAP - 1);
  slotA[t] = s;
  if (!keep) gvArr[t] = 0.f;
  else inv[e * C_CAP + pos] = t;
}

// ---------------------------------------------------------------- dispatch (fp32 -> bf16 scatter)
__global__ __launch_bounds__(256)
void moe455_dispatch(const float* __restrict__ x, const int* __restrict__ idxArr,
                     const int* __restrict__ slotA, const float* __restrict__ gvArr,
                     __bf16* __restrict__ disp) {
  int wid = threadIdx.x >> 5, lane = threadIdx.x & 31;
  int t = blockIdx.x * 8 + wid;
  if (gvArr[t] == 0.0f) return;  // dropped token contributes zeros (buffer pre-zeroed)
  int e = idxArr[t], s = slotA[t];
  const float2* x2 = reinterpret_cast<const float2*>(x + (size_t)t * D_DIM);
  unsigned* d2 = reinterpret_cast<unsigned*>(disp + ((size_t)e * C_CAP + s) * D_DIM);
#pragma unroll 4
  for (int i = 0; i < D_DIM / 64; ++i) {
    int d = lane + i * 32;
    float2 f = x2[d];
    d2[d] = (unsigned)f2bfu(f.x) | ((unsigned)f2bfu(f.y) << 16);
  }
}

// ---------------------------------------------------------------- expert GEMMs (WMMA bf16)
// 128x128 block tile, 256 threads = 8 waves; wave: 32x64 -> 2x4 16x16 accum tiles.
// A [C_CAP x KDIM] bf16 row-major per expert; WT [NDIM x KDIM] bf16 per expert (pre-transposed),
// so A and B tiles are identical 128x32 copies done with async global->LDS + double buffering.
template <int KDIM, int NDIM, bool FFN1>
__global__ __launch_bounds__(256)
void moe455_ffn(const __bf16* __restrict__ Abase, const __bf16* __restrict__ WTbase,
                __bf16* __restrict__ Hout, float* __restrict__ Out,
                const int* __restrict__ inv, const float* __restrict__ gv) {
  constexpr int BM = 128, BN = 128, BK = 32;
  __shared__ alignas(64) __bf16 aLds[2][BM][BK];
  __shared__ alignas(64) __bf16 bLds[2][BN][BK];

  const int e  = blockIdx.z;
  const int m0 = blockIdx.y * BM;
  const int n0 = blockIdx.x * BN;
  const int tid = threadIdx.x;
  const int wid = tid >> 5, lane = tid & 31;
  const int lrow = lane & 15, lhalf = lane >> 4;
  const int wm = (wid & 3) * 32;
  const int wn = (wid >> 2) * 64;

  const __bf16* A  = Abase  + (size_t)e * C_CAP * KDIM + (size_t)m0 * KDIM;
  const __bf16* BW = WTbase + (size_t)e * NDIM * KDIM + (size_t)n0 * KDIM;

  // per-thread tile-copy coords: 2 chunks of 8 bf16 (16B) per matrix
  const int r0 = tid >> 2, c0 = (tid & 3) * 8;        // chunk 0: rows 0..63
  const int r1 = r0 + 64;                              // chunk 1: rows 64..127

  auto issue_tiles = [&](int k0, int buf) {
#pragma unroll
    for (int half = 0; half < 2; ++half) {
      int r = half ? r1 : r0;
      unsigned la = (unsigned)(uintptr_t)&aLds[buf][r][c0];
      unsigned long long ga = (unsigned long long)(A + (size_t)r * KDIM + k0 + c0);
      asm volatile("global_load_async_to_lds_b128 %0, %1, off"
                   :: "v"(la), "v"(ga) : "memory");
      unsigned lb = (unsigned)(uintptr_t)&bLds[buf][r][c0];
      unsigned long long gb = (unsigned long long)(BW + (size_t)r * KDIM + k0 + c0);
      asm volatile("global_load_async_to_lds_b128 %0, %1, off"
                   :: "v"(lb), "v"(gb) : "memory");
    }
  };

  v8f acc[2][4];
#pragma unroll
  for (int i = 0; i < 2; ++i)
#pragma unroll
    for (int j = 0; j < 4; ++j)
#pragma unroll
      for (int v = 0; v < 8; ++v) acc[i][j][v] = 0.f;

  issue_tiles(0, 0);
  asm volatile("s_wait_asynccnt 0" ::: "memory");
  __syncthreads();

  int cur = 0;
  for (int k0 = 0; k0 < KDIM; k0 += BK) {
    if (k0 + BK < KDIM) issue_tiles(k0 + BK, cur ^ 1);  // prefetch next tile (async)

    // A fragment: lane half h holds K = h*16 .. h*16+15 contiguous (matches B)
    v16bf af[2];
#pragma unroll
    for (int i = 0; i < 2; ++i) {
      const __bf16* rp = &aLds[cur][wm + i * 16 + lrow][0];
      v8bf lo = *reinterpret_cast<const v8bf*>(rp + lhalf * 16);
      v8bf hi = *reinterpret_cast<const v8bf*>(rp + lhalf * 16 + 8);
      af[i] = __builtin_shufflevector(lo, hi, 0, 1, 2, 3, 4, 5, 6, 7,
                                      8, 9, 10, 11, 12, 13, 14, 15);
    }
    // B fragment: lane holds column n = lane%16, K = lhalf*16 .. +15 contiguous
    v16bf bfr[4];
#pragma unroll
    for (int j = 0; j < 4; ++j)
      bfr[j] = *reinterpret_cast<const v16bf*>(&bLds[cur][wn + j * 16 + lrow][lhalf * 16]);

#pragma unroll
    for (int i = 0; i < 2; ++i)
#pragma unroll
      for (int j = 0; j < 4; ++j)
        acc[i][j] = __builtin_amdgcn_wmma_f32_16x16x32_bf16(
            false, af[i], false, bfr[j], (short)0, acc[i][j], false, false);

    asm volatile("s_wait_asynccnt 0" ::: "memory");   // next tile landed
    __syncthreads();                                  // everyone done with `cur`
    cur ^= 1;
  }

  // C/D layout: elem v of lane -> M = tileM + v + 8*lhalf, N = tileN + lane%16
  if constexpr (FFN1) {
    __bf16* H = Hout + (size_t)e * C_CAP * NDIM;
#pragma unroll
    for (int i = 0; i < 2; ++i)
#pragma unroll
      for (int v = 0; v < 8; ++v) {
        int r = m0 + wm + i * 16 + v + 8 * lhalf;
        __bf16* hr = H + (size_t)r * NDIM + n0 + wn;
#pragma unroll
        for (int j = 0; j < 4; ++j)
          hr[j * 16 + lrow] = f2bf(gelu_tanh(acc[i][j][v]));
      }
  } else {
#pragma unroll
    for (int i = 0; i < 2; ++i)
#pragma unroll
      for (int v = 0; v < 8; ++v) {
        int r = m0 + wm + i * 16 + v + 8 * lhalf;
        int tok = inv[e * C_CAP + r];
        if (tok >= 0) {
          float g = gv[tok];
          float* orow = Out + (size_t)tok * NDIM + n0 + wn;
#pragma unroll
          for (int j = 0; j < 4; ++j)
            orow[j * 16 + lrow] = acc[i][j][v] * g;
        }
      }
  }
}

// ---------------------------------------------------------------- launch
extern "C" void kernel_launch(void* const* d_in, const int* in_sizes, int n_in,
                              void* d_out, int out_size, void* d_ws, size_t ws_size,
                              hipStream_t stream) {
  (void)in_sizes; (void)n_in; (void)out_size; (void)ws_size;
  const float* x  = (const float*)d_in[0];
  const float* wg = (const float*)d_in[1];
  const float* w1 = (const float*)d_in[2];
  const float* w2 = (const float*)d_in[3];
  float* out  = (float*)d_out;
  float* laux = out + (size_t)T_TOK * D_DIM;

  char* ws = (char*)d_ws;
  size_t off = 0;
  auto take = [&](size_t bytes) { size_t r = off; off += (bytes + 255) & ~(size_t)255; return r; };
  __bf16* disp   = (__bf16*)(ws + take((size_t)E_EXP * C_CAP * D_DIM * 2));
  __bf16* hbuf   = (__bf16*)(ws + take((size_t)E_EXP * C_CAP * DFF_DIM * 2));
  __bf16* w1T    = (__bf16*)(ws + take((size_t)E_EXP * D_DIM * DFF_DIM * 2));
  __bf16* w2T    = (__bf16*)(ws + take((size_t)E_EXP * DFF_DIM * D_DIM * 2));
  int*    idxArr = (int*)   (ws + take((size_t)T_TOK * 4));
  int*    slotA  = (int*)   (ws + take((size_t)T_TOK * 4));
  float*  gvArr  = (float*) (ws + take((size_t)T_TOK * 4));
  int*    inv    = (int*)   (ws + take((size_t)E_EXP * C_CAP * 4));
  int*    hist   = (int*)   (ws + take((size_t)NBLK * E_EXP * 4));
  int*    boff   = (int*)   (ws + take((size_t)NBLK * E_EXP * 4));
  float*  gsum   = (float*) (ws + take((size_t)E_EXP * 4));

  moe455_fill_u32<<<1024, 256, 0, stream>>>((unsigned*)disp, 0u, (long)E_EXP * C_CAP * D_DIM / 2);
  moe455_fill_u32<<<1024, 256, 0, stream>>>((unsigned*)out, 0u, (long)T_TOK * D_DIM);
  moe455_fill_u32<<<64, 256, 0, stream>>>((unsigned*)inv, 0xFFFFFFFFu, (long)E_EXP * C_CAP);
  moe455_fill_u32<<<1, 64, 0, stream>>>((unsigned*)gsum, 0u, (long)E_EXP);

  // one-time weight convert+transpose: [E][K][N] f32 -> [E][N][K] bf16
  moe455_wcvt<D_DIM, DFF_DIM>
      <<<dim3(DFF_DIM / 64, D_DIM / 64, E_EXP), 256, 0, stream>>>(w1, w1T);
  moe455_wcvt<DFF_DIM, D_DIM>
      <<<dim3(D_DIM / 64, DFF_DIM / 64, E_EXP), 256, 0, stream>>>(w2, w2T);

  moe455_gate<<<T_TOK / 8, 256, 0, stream>>>(x, wg, idxArr, gvArr, gsum);
  moe455_hist<<<NBLK, 256, 0, stream>>>(idxArr, hist);
  moe455_scan<<<1, 32, 0, stream>>>(hist, boff, gsum, laux);
  moe455_rank<<<NBLK, 256, 0, stream>>>(idxArr, boff, slotA, gvArr, inv);
  moe455_dispatch<<<T_TOK / 8, 256, 0, stream>>>(x, idxArr, slotA, gvArr, disp);

  moe455_ffn<D_DIM, DFF_DIM, true>
      <<<dim3(DFF_DIM / 128, C_CAP / 128, E_EXP), 256, 0, stream>>>(
          disp, w1T, hbuf, nullptr, nullptr, nullptr);
  moe455_ffn<DFF_DIM, D_DIM, false>
      <<<dim3(D_DIM / 128, C_CAP / 128, E_EXP), 256, 0, stream>>>(
          hbuf, w2T, nullptr, out, inv, gvArr);
}
